// GAT_38482906972424
// MI455X (gfx1250) — matrix operand order repaired and checked by reading, not compiled, not run on hardware
//
#include <hip/hip_runtime.h>

#define IN_DIM   256
#define HID      64
#define HEADS    4
#define OUT_DIM  256
#define NEG_SLOPE 0.01f

typedef __attribute__((ext_vector_type(16))) _Float16 v16h;
typedef __attribute__((ext_vector_type(8)))  float    v8f;

// ---- helpers -------------------------------------------------------------
__device__ __forceinline__ unsigned flipF(float f) {
    unsigned u = __float_as_uint(f);
    return (u & 0x80000000u) ? ~u : (u | 0x80000000u);   // monotone order-preserving map
}
__device__ __forceinline__ float unflipF(unsigned u) {
    if (u == 0u) return 0.0f;                            // "no edge" sentinel (ref: isfinite->0)
    return __uint_as_float((u & 0x80000000u) ? (u ^ 0x80000000u) : ~u);
}
__device__ __forceinline__ float leakyf(float x) { return x > 0.f ? x : NEG_SLOPE * x; }

// ---- f32 -> f16 convert --------------------------------------------------
__global__ void cvt_f16_kernel(const float* __restrict__ in, _Float16* __restrict__ out, int n) {
    int i = blockIdx.x * blockDim.x + threadIdx.x;
    if (i < n) out[i] = (_Float16)in[i];
}

// ---- WMMA GEMM: C[M x 256] = A[M x 256] * W[256 x 256]^T -----------------
// One block = 8 waves; each wave computes two 16x16 tiles of one 16-row band.
// A-matrix 16x32 f16 lane layout: lane m%16 holds row m; element e -> K = e + 8*hi (+8 if e>=8; see doc)
// per CDNA5 ISA: lanes 0-15 VGPR0..3 K=0..7, VGPR4..7 K=16..23; lanes 16-31 +8.
__global__ void __launch_bounds__(256)
gemm_f16_wmma_kernel(const _Float16* __restrict__ A,
                     const _Float16* __restrict__ W,
                     float* __restrict__ C, int M) {
    const int K = 256;
    const int lane = threadIdx.x & 31;
    const int wave = threadIdx.x >> 5;
    const int hi   = (lane >> 4) & 1;
    const int l16  = lane & 15;
    const int tileM = blockIdx.x << 4;
    const int n0 = (wave * 2) << 4;       // this wave's first output-column tile
    const int n1 = n0 + 16;

    v8f acc0 = {}; v8f acc1 = {};
    const _Float16* Arow = A + (size_t)(tileM + l16) * K + 8 * hi;
    const _Float16* B0   = W + (size_t)(n0 + l16) * K + 8 * hi;
    const _Float16* B1   = W + (size_t)(n1 + l16) * K + 8 * hi;

    for (int k0 = 0; k0 < K; k0 += 32) {
        union { v16h v; float2 f2[4]; float4 f4[2]; } a, b0, b1;
        // A: elements 0..7 -> K = k0+8hi+0..7 ; elements 8..15 -> K = k0+8hi+16..23
        a.f2[0] = *(const float2*)(Arow + k0);
        a.f2[1] = *(const float2*)(Arow + k0 + 4);
        a.f2[2] = *(const float2*)(Arow + k0 + 16);
        a.f2[3] = *(const float2*)(Arow + k0 + 20);
        // B: lane holds column n; elements 0..7 -> K = k0+8hi+0..7 ; elements 8..15 -> K = k0+8hi+16..23
        b0.f4[0] = *(const float4*)(B0 + k0);
        b0.f4[1] = *(const float4*)(B0 + k0 + 16);
        b1.f4[0] = *(const float4*)(B1 + k0);
        b1.f4[1] = *(const float4*)(B1 + k0 + 16);
        acc0 = __builtin_amdgcn_wmma_f32_16x16x32_f16(false, a.v, false, b0.v,
                                                      (short)0, acc0, false, false);
        acc1 = __builtin_amdgcn_wmma_f32_16x16x32_f16(false, a.v, false, b1.v,
                                                      (short)0, acc1, false, false);
    }
    // C/D layout: VGPR i -> row tileM + i + 8*hi, col = tileN + (lane&15)
#pragma unroll
    for (int i = 0; i < 8; ++i) {
        int row = tileM + i + 8 * hi;
        C[(size_t)row * 256 + n0 + l16] = acc0[i];
        C[(size_t)row * 256 + n1 + l16] = acc1[i];
    }
}

// ---- layer-1 attention scores: es/ed[n,h] = dot(z[n,h,:], a1[h, :64 / 64:]) ----
__global__ void scores1_kernel(const float* __restrict__ z, const float* __restrict__ a1,
                               float* __restrict__ es, float* __restrict__ ed, int N) {
    int t = blockIdx.x * blockDim.x + threadIdx.x;
    if (t >= N * HEADS) return;
    int n = t >> 2, hd = t & (HEADS - 1);
    const float* zp = z + (size_t)n * 256 + hd * HID;
    const float* as = a1 + hd * 2 * HID;
    const float* ad = as + HID;
    float s = 0.f, d = 0.f;
#pragma unroll 4
    for (int o = 0; o < HID; ++o) { float zv = zp[o]; s += zv * as[o]; d += zv * ad[o]; }
    es[t] = s; ed[t] = d;
}

// ---- layer-2 scores: s_src/s_dst[n] = dot(z2[n,:], a2[:256] / a2[256:]) ----
__global__ void scores2_kernel(const float* __restrict__ z2, const float* __restrict__ a2,
                               float* __restrict__ ss, float* __restrict__ sd, int N) {
    int n = blockIdx.x * blockDim.x + threadIdx.x;
    if (n >= N) return;
    const float* zp = z2 + (size_t)n * 256;
    float s = 0.f, d = 0.f;
#pragma unroll 4
    for (int i = 0; i < 256; ++i) { float zv = zp[i]; s += zv * a2[i]; d += zv * a2[256 + i]; }
    ss[n] = s; sd[n] = d;
}

// ---- segment softmax pass 1: atomic max of leaky(es[src]+ed[dst]) --------
__global__ void edge_max_kernel(const int* __restrict__ src, const int* __restrict__ dst,
                                const float* __restrict__ es, const float* __restrict__ ed,
                                unsigned* __restrict__ mx, int E, int H) {
    int e = blockIdx.x * blockDim.x + threadIdx.x;
    if (e >= E) return;
    int s = src[e], d = dst[e];
    for (int h = 0; h < H; ++h) {
        float v = leakyf(es[s * H + h] + ed[d * H + h]);
        atomicMax(&mx[d * H + h], flipF(v));
    }
}

// ---- pass 2: ex = exp(e - max); stash ex; atomicAdd segment sums ---------
__global__ void edge_expsum_kernel(const int* __restrict__ src, const int* __restrict__ dst,
                                   const float* __restrict__ es, const float* __restrict__ ed,
                                   const unsigned* __restrict__ mx, float* __restrict__ ssum,
                                   float* __restrict__ exbuf, int E, int H) {
    int e = blockIdx.x * blockDim.x + threadIdx.x;
    if (e >= E) return;
    int s = src[e], d = dst[e];
    for (int h = 0; h < H; ++h) {
        float v = leakyf(es[s * H + h] + ed[d * H + h]);
        float ex = __expf(v - unflipF(mx[d * H + h]));
        exbuf[(size_t)e * H + h] = ex;
        atomicAdd(&ssum[d * H + h], ex);
    }
}

// ---- pass 3: acc[dst, c] += (ex/sum) * feat[src, c]  (one block per edge) ----
__global__ void edge_accum_kernel(const int* __restrict__ src, const int* __restrict__ dst,
                                  const float* __restrict__ exbuf, const float* __restrict__ ssum,
                                  const float* __restrict__ feat, float* __restrict__ acc,
                                  int H) {
    int e = blockIdx.x;
    int c = threadIdx.x;                         // 256 channels
    int s = src[e], d = dst[e];
    int h = (H == 1) ? 0 : (c >> 6);             // 256/4 = 64 channels per head
    float coef = exbuf[(size_t)e * H + h] / ssum[d * H + h];
    atomicAdd(&acc[(size_t)d * 256 + c], coef * feat[(size_t)s * 256 + c]);
}

// ---- ELU + convert to f16 for layer-2 GEMM -------------------------------
__global__ void elu_cvt_kernel(const float* __restrict__ m, _Float16* __restrict__ out16, int n) {
    int i = blockIdx.x * blockDim.x + threadIdx.x;
    if (i >= n) return;
    float v = m[i];
    v = v > 0.f ? v : (__expf(v) - 1.0f);
    out16[i] = (_Float16)v;
}

extern "C" void kernel_launch(void* const* d_in, const int* in_sizes, int n_in,
                              void* d_out, int out_size, void* d_ws, size_t ws_size,
                              hipStream_t stream) {
    const float* hin = (const float*)d_in[0];
    const float* W1  = (const float*)d_in[1];
    const float* a1  = (const float*)d_in[2];
    const float* W2  = (const float*)d_in[3];
    const float* a2  = (const float*)d_in[4];
    const int*   src = (const int*)d_in[5];
    const int*   dst = (const int*)d_in[6];
    float* out = (float*)d_out;

    const int N = in_sizes[0] / IN_DIM;     // 50000 (divisible by 16)
    const int E = in_sizes[5];              // 850000

    // ---- workspace carve-out (~150 MB) ----
    char* w = (char*)d_ws;
    auto alloc = [&](size_t b) -> void* {
        void* p = (void*)w; w += (b + 255) & ~(size_t)255; return p;
    };
    _Float16* hf16 = (_Float16*)alloc((size_t)N * 256 * 2);  // reused as h1f16
    _Float16* W1f  = (_Float16*)alloc((size_t)256 * 256 * 2);
    _Float16* W2f  = (_Float16*)alloc((size_t)256 * 256 * 2);
    float*    z    = (float*)alloc((size_t)N * 256 * 4);     // reused as z2
    float*    es   = (float*)alloc((size_t)N * HEADS * 4);
    float*    ed   = (float*)alloc((size_t)N * HEADS * 4);
    unsigned* mx1  = (unsigned*)alloc((size_t)N * HEADS * 4);
    float*    ss1  = (float*)alloc((size_t)N * HEADS * 4);
    float*    ex1  = (float*)alloc((size_t)E * HEADS * 4);
    float*    m    = (float*)alloc((size_t)N * 256 * 4);
    float*    s2s  = (float*)alloc((size_t)N * 4);
    float*    s2d  = (float*)alloc((size_t)N * 4);
    unsigned* mx2  = (unsigned*)alloc((size_t)N * 4);
    float*    ss2  = (float*)alloc((size_t)N * 4);
    float*    ex2  = (float*)alloc((size_t)E * 4);

    // ---- zero accumulators (graph-capture safe) ----
    hipMemsetAsync(mx1, 0, (size_t)N * HEADS * 4, stream);
    hipMemsetAsync(ss1, 0, (size_t)N * HEADS * 4, stream);
    hipMemsetAsync(m,   0, (size_t)N * 256 * 4,   stream);
    hipMemsetAsync(mx2, 0, (size_t)N * 4,         stream);
    hipMemsetAsync(ss2, 0, (size_t)N * 4,         stream);
    hipMemsetAsync(out, 0, (size_t)N * 256 * 4,   stream);

    const int T = 256;
    // convert inputs to f16 for WMMA
    cvt_f16_kernel<<<(N * 256 + T - 1) / T, T, 0, stream>>>(hin, hf16, N * 256);
    cvt_f16_kernel<<<(65536 + T - 1) / T,   T, 0, stream>>>(W1, W1f, 65536);
    cvt_f16_kernel<<<(65536 + T - 1) / T,   T, 0, stream>>>(W2, W2f, 65536);

    // ---- layer 1: z = h @ W1^T  (WMMA) ----
    gemm_f16_wmma_kernel<<<N / 16, 256, 0, stream>>>(hf16, W1f, z, N);
    scores1_kernel<<<(N * HEADS + T - 1) / T, T, 0, stream>>>(z, a1, es, ed, N);
    edge_max_kernel   <<<(E + T - 1) / T, T, 0, stream>>>(src, dst, es, ed, mx1, E, HEADS);
    edge_expsum_kernel<<<(E + T - 1) / T, T, 0, stream>>>(src, dst, es, ed, mx1, ss1, ex1, E, HEADS);
    edge_accum_kernel <<<E, 256, 0, stream>>>(src, dst, ex1, ss1, z, m, HEADS);

    // ---- ELU + layer 2: z2 = elu(m) @ W2^T (WMMA) ----
    elu_cvt_kernel<<<(N * 256 + T - 1) / T, T, 0, stream>>>(m, hf16, N * 256);
    gemm_f16_wmma_kernel<<<N / 16, 256, 0, stream>>>(hf16, W2f, z, N);   // z := z2
    scores2_kernel<<<(N + T - 1) / T, T, 0, stream>>>(z, a2, s2s, s2d, N);
    edge_max_kernel   <<<(E + T - 1) / T, T, 0, stream>>>(src, dst, s2s, s2d, mx2, E, 1);
    edge_expsum_kernel<<<(E + T - 1) / T, T, 0, stream>>>(src, dst, s2s, s2d, mx2, ss2, ex2, E, 1);
    edge_accum_kernel <<<E, 256, 0, stream>>>(src, dst, ex2, ss2, z, out, 1);
}
